// ALIGNN_35527969473184
// MI455X (gfx1250) — compile-verified, matrix operand order checked by reading, not run on hardware
//
#include <hip/hip_runtime.h>
#include <hip/hip_bf16.h>

// ---------------------------------------------------------------------------
// ALIGNN forward on MI455X (gfx1250, wave32).
//  - Dense linears: weight-stationary bf16 WMMA (v_wmma_f32_16x16x32_bf16,
//    f32 accumulate). K-extent templated -> fully unrolled WMMA chains.
//  - Segment sums via f32 L2 atomics (accumulation tables fit in 192MB L2).
//  - Wave-per-row LayerNorm with __shfl_xor reductions (wave32).
// ---------------------------------------------------------------------------

typedef __attribute__((ext_vector_type(16))) __bf16 v16bf;
typedef __attribute__((ext_vector_type(8)))  float  v8f;

#define HID 128

__device__ __forceinline__ float sigmoidf_(float x) { return 1.0f / (1.0f + __expf(-x)); }
__device__ __forceinline__ float siluf_(float x)    { return x * sigmoidf_(x); }

// Load 16 f32 in the WMMA 16-bit A/B per-lane K pattern and convert to bf16.
// Elements 0..7  = K[kb+0 .. kb+7], elements 8..15 = K[kb+16 .. kb+23].
__device__ __forceinline__ v16bf cvt16_fast(const float* __restrict__ p) {
    const float4* q = (const float4*)p;
    float4 t0 = q[0], t1 = q[1], t2 = q[4], t3 = q[5];
    v16bf r;
    r[0]=(__bf16)t0.x;  r[1]=(__bf16)t0.y;  r[2]=(__bf16)t0.z;  r[3]=(__bf16)t0.w;
    r[4]=(__bf16)t1.x;  r[5]=(__bf16)t1.y;  r[6]=(__bf16)t1.z;  r[7]=(__bf16)t1.w;
    r[8]=(__bf16)t2.x;  r[9]=(__bf16)t2.y;  r[10]=(__bf16)t2.z; r[11]=(__bf16)t2.w;
    r[12]=(__bf16)t3.x; r[13]=(__bf16)t3.y; r[14]=(__bf16)t3.z; r[15]=(__bf16)t3.w;
    return r;
}

__device__ __forceinline__ v16bf cvt16_tail(const float* __restrict__ row, int kb, int din) {
    v16bf r;
#pragma unroll
    for (int i = 0; i < 16; i++) {
        int k = kb + (i < 8 ? i : i + 8);
        float f = (k < din) ? row[k] : 0.0f;
        r[i] = (__bf16)f;
    }
    return r;
}

// out[M,dout] = X[M,DIN] @ W[dout,DIN]^T + bias.
// Weight-stationary: blockIdx.y selects a 16-wide column block; each wave
// preloads its full-K B fragment into registers once, then streams M-tiles.
// All waves keep EXEC all-ones (no early-outs; zero-trip loops are uniform).
template <int DIN>
__global__ __launch_bounds__(256) void gemm_ws_wmma(
    const float* __restrict__ X, const float* __restrict__ W,
    const float* __restrict__ bias, float* __restrict__ out,
    int Mrows, int dout)
{
    constexpr int KSTEPS = (DIN + 31) / 32;
    int lane = threadIdx.x & 31;
    int l15  = lane & 15;
    int hi   = lane >> 4;                    // K-half select (0 or 1)
    int nt   = blockIdx.y;                   // column tile (dout/16 of them)
    int wavesPerCol = gridDim.x * 8;
    int wcol = blockIdx.x * 8 + (threadIdx.x >> 5);
    int ntm  = Mrows >> 4;
    int col  = nt * 16 + l15;                // B-matrix column for this lane

    // Preload the whole K-extent of B (weights) into registers: KSTEPS x 8 VGPRs.
    const float* wrow = W + (size_t)col * DIN;
    v16bf bf[KSTEPS];
#pragma unroll
    for (int ks = 0; ks < KSTEPS; ks++) {
        int kb = ks * 32 + hi * 8;
        if (ks * 32 + 32 <= DIN) bf[ks] = cvt16_fast(wrow + kb);
        else                     bf[ks] = cvt16_tail(wrow, kb, DIN);
    }
    float bv = bias[col];                    // C/D: all 8 regs are column `col`

    for (int mt = wcol; mt < ntm; mt += wavesPerCol) {
        const float* xrow = X + (size_t)(mt * 16 + l15) * DIN;
        v8f acc;
#pragma unroll
        for (int i = 0; i < 8; i++) acc[i] = bv;
#pragma unroll
        for (int ks = 0; ks < KSTEPS; ks++) {
            int kb = ks * 32 + hi * 8;
            v16bf a;
            if (ks * 32 + 32 <= DIN) a = cvt16_fast(xrow + kb);
            else                     a = cvt16_tail(xrow, kb, DIN);
            acc = __builtin_amdgcn_wmma_f32_16x16x32_bf16(
                      false, a, false, bf[ks], (short)0, acc, false, false);
        }
        // D layout: acc[r] -> row mt*16 + r + 8*hi, column col
        float* orow = out + (size_t)(mt * 16 + hi * 8) * dout + (size_t)nt * 16 + l15;
#pragma unroll
        for (int r8 = 0; r8 < 8; r8++) orow[(size_t)r8 * dout] = acc[r8];
    }
}

// In-place LayerNorm + SiLU over last dim d (64 or 128); one wave per row.
__global__ __launch_bounds__(256) void ln_silu_k(
    float* __restrict__ X, const float* __restrict__ g, const float* __restrict__ b,
    int rows, int d)
{
    int wave = (blockIdx.x * 256 + threadIdx.x) >> 5;
    int lane = threadIdx.x & 31;
    if (wave >= rows) return;
    float* row = X + (size_t)wave * d;
    int per = d >> 5;                       // 2 (d=64) or 4 (d=128)
    float v[4];
    float s = 0.f;
    for (int i = 0; i < per; i++) { v[i] = row[lane + 32 * i]; s += v[i]; }
    for (int off = 16; off > 0; off >>= 1) s += __shfl_xor(s, off, 32);
    float mu = s / (float)d;
    float var = 0.f;
    for (int i = 0; i < per; i++) { float dl = v[i] - mu; var += dl * dl; }
    for (int off = 16; off > 0; off >>= 1) var += __shfl_xor(var, off, 32);
    float inv = rsqrtf(var / (float)d + 1e-5f);
    for (int i = 0; i < per; i++) {
        int idx = lane + 32 * i;
        float t = (v[i] - mu) * inv * g[idx] + b[idx];
        row[idx] = siluf_(t);
    }
}

// bondlength + 80-bin RBF over [0,6]
__global__ __launch_bounds__(256) void bond_rbf_k(
    const float* __restrict__ r, float* __restrict__ out, int E)
{
    int tid = blockIdx.x * 256 + threadIdx.x;
    if (tid >= E * 80) return;
    int e = tid / 80;
    int i = tid - e * 80;
    float rx = r[e * 3 + 0], ry = r[e * 3 + 1], rz = r[e * 3 + 2];
    float bl = sqrtf(rx * rx + ry * ry + rz * rz);
    float c = (6.0f / 79.0f) * (float)i;
    float dl = bl - c;
    out[tid] = __expf(-(79.0f / 6.0f) * dl * dl);
}

// 40-bin RBF over [-1,1]
__global__ __launch_bounds__(256) void angle_rbf_k(
    const float* __restrict__ x, float* __restrict__ out, int E)
{
    int tid = blockIdx.x * 256 + threadIdx.x;
    if (tid >= E * 40) return;
    int e = tid / 40;
    int i = tid - e * 40;
    float c = -1.0f + (2.0f / 39.0f) * (float)i;
    float dl = x[e] - c;
    out[tid] = __expf(-(39.0f / 2.0f) * dl * dl);
}

// M[e,:] += A[src[e],:] + B[dst[e],:]   (M pre-filled with edge_gate(y))
__global__ __launch_bounds__(256) void egc_message_k(
    float* __restrict__ M, const float* __restrict__ A, const float* __restrict__ B,
    const int* __restrict__ src, const int* __restrict__ dst, int E)
{
    int wave = (blockIdx.x * 256 + threadIdx.x) >> 5;
    int lane = threadIdx.x & 31;
    if (wave >= E) return;
    int s = src[wave], t = dst[wave];
    const float4* ap = (const float4*)(A + (size_t)s * HID) + lane;
    const float4* bp = (const float4*)(B + (size_t)t * HID) + lane;
    float4* mp = (float4*)(M + (size_t)wave * HID) + lane;
    float4 a = *ap, b = *bp, m = *mp;
    m.x += a.x + b.x; m.y += a.y + b.y; m.z += a.z + b.z; m.w += a.w + b.w;
    *mp = m;
}

// sigma = sigmoid(m); SUMH[dst] += sigma*Bh[src]; SUMS[dst] += sigma  (L2 atomics)
__global__ __launch_bounds__(256) void egc_scatter_k(
    float* __restrict__ SUMH, float* __restrict__ SUMS,
    const float* __restrict__ M, const float* __restrict__ BH,
    const int* __restrict__ src, const int* __restrict__ dst, int E)
{
    int wave = (blockIdx.x * 256 + threadIdx.x) >> 5;
    int lane = threadIdx.x & 31;
    if (wave >= E) return;
    int s = src[wave], t = dst[wave];
    const float* mrow = M  + (size_t)wave * HID + lane * 4;
    const float* bh   = BH + (size_t)s    * HID + lane * 4;
    float* sh = SUMH + (size_t)t * HID + lane * 4;
    float* ss = SUMS + (size_t)t * HID + lane * 4;
#pragma unroll
    for (int i = 0; i < 4; i++) {
        float sg = sigmoidf_(mrow[i]);
        atomicAdd(sh + i, sg * bh[i]);
        atomicAdd(ss + i, sg);
    }
}

// x += silu(bn(Au + sumh/(sums+1e-6)))
__global__ __launch_bounds__(256) void egc_node_update_k(
    float* __restrict__ X, const float* __restrict__ AU,
    const float* __restrict__ SUMH, const float* __restrict__ SUMS,
    const float* __restrict__ g, const float* __restrict__ b, int total)
{
    int idx = blockIdx.x * 256 + threadIdx.x;
    if (idx >= total) return;
    int f = idx & (HID - 1);
    float h = SUMH[idx] / (SUMS[idx] + 1e-6f);
    float t = AU[idx] + h;
    t = t * (g[f] * rsqrtf(1.0f + 1e-5f)) + b[f];
    X[idx] += siluf_(t);
}

// y += silu(bn(m))
__global__ __launch_bounds__(256) void egc_edge_update_k(
    float* __restrict__ Y, const float* __restrict__ M,
    const float* __restrict__ g, const float* __restrict__ b, int total)
{
    int idx = blockIdx.x * 256 + threadIdx.x;
    if (idx >= total) return;
    int f = idx & (HID - 1);
    float t = M[idx] * (g[f] * rsqrtf(1.0f + 1e-5f)) + b[f];
    Y[idx] += siluf_(t);
}

// mean-pool nodes then fc -> scalar
__global__ void readout_k(const float* __restrict__ X, const float* __restrict__ fcw,
                          const float* __restrict__ fcb, float* __restrict__ out, int n)
{
    __shared__ float h[HID];
    int f = threadIdx.x;            // 128 threads
    float s = 0.f;
    for (int m = 0; m < n; m++) s += X[(size_t)m * HID + f];
    h[f] = s / (float)n;
    __syncthreads();
    if (f == 0) {
        float o = fcb[0];
        for (int i = 0; i < HID; i++) o += fcw[i] * h[i];
        out[0] = o;
    }
}

// ---------------------------------------------------------------------------
// Host side
// ---------------------------------------------------------------------------

namespace {

struct Egc {
    const float *bn_e_b, *bn_e_g, *bn_n_b, *bn_n_g;
    const float *dg_b, *dg_w;   // dst_gate
    const float *du_b, *du_w;   // dst_update (Bh)
    const float *eg_b, *eg_w;   // edge_gate
    const float *sg_b, *sg_w;   // src_gate
    const float *su_b, *su_w;   // src_update (Au)
};

Egc egc_at(void* const* d_in, int base) {
    Egc p;
    p.bn_e_b = (const float*)d_in[base + 0];  p.bn_e_g = (const float*)d_in[base + 1];
    p.bn_n_b = (const float*)d_in[base + 2];  p.bn_n_g = (const float*)d_in[base + 3];
    p.dg_b = (const float*)d_in[base + 4];    p.dg_w = (const float*)d_in[base + 5];
    p.du_b = (const float*)d_in[base + 6];    p.du_w = (const float*)d_in[base + 7];
    p.eg_b = (const float*)d_in[base + 8];    p.eg_w = (const float*)d_in[base + 9];
    p.sg_b = (const float*)d_in[base + 10];   p.sg_w = (const float*)d_in[base + 11];
    p.su_b = (const float*)d_in[base + 12];   p.su_w = (const float*)d_in[base + 13];
    return p;
}

struct Mlp { const float *b, *ln_b, *ln_g, *w; };
Mlp mlp_at(void* const* d_in, int base) {
    Mlp m;
    m.b    = (const float*)d_in[base + 0];
    m.ln_b = (const float*)d_in[base + 1];
    m.ln_g = (const float*)d_in[base + 2];
    m.w    = (const float*)d_in[base + 3];
    return m;
}

void launch_gemm(float* out, const float* X, const float* W, const float* b,
                 int M, int din, int dout, hipStream_t s) {
    int ntm = M >> 4;
    int ntn = dout >> 4;
    int bx = (ntm + 7) / 8;
    if (bx > 128) bx = 128;          // ~8K waves total at dout=128: fills the chip
    dim3 grid(bx, ntn);
    switch (din) {
    case 128: gemm_ws_wmma<128><<<grid, 256, 0, s>>>(X, W, b, out, M, dout); break;
    case  92: gemm_ws_wmma< 92><<<grid, 256, 0, s>>>(X, W, b, out, M, dout); break;
    case  80: gemm_ws_wmma< 80><<<grid, 256, 0, s>>>(X, W, b, out, M, dout); break;
    case  64: gemm_ws_wmma< 64><<<grid, 256, 0, s>>>(X, W, b, out, M, dout); break;
    case  40: gemm_ws_wmma< 40><<<grid, 256, 0, s>>>(X, W, b, out, M, dout); break;
    default: break; // unused
    }
}

void launch_ln(float* X, const float* g, const float* b, int rows, int d, hipStream_t s) {
    int blocks = (rows * 32 + 255) / 256;
    ln_silu_k<<<blocks, 256, 0, s>>>(X, g, b, rows, d);
}

struct Ws {
    float *X, *Y, *Z, *Mb, *G0, *G1, *G2, *G3, *SH, *SS;
};

void run_egc(const Egc& p, const int* src, const int* dst,
             float* x, float* y, int n, int E, const Ws& w, hipStream_t s) {
    launch_gemm(w.G0, x, p.sg_w, p.sg_b, n, HID, HID, s);   // src_gate
    launch_gemm(w.G1, x, p.dg_w, p.dg_b, n, HID, HID, s);   // dst_gate
    launch_gemm(w.G2, x, p.du_w, p.du_b, n, HID, HID, s);   // Bh = dst_update
    launch_gemm(w.G3, x, p.su_w, p.su_b, n, HID, HID, s);   // Au = src_update
    launch_gemm(w.Mb, y, p.eg_w, p.eg_b, E, HID, HID, s);   // edge_gate(y)
    {
        int blocks = (E * 32 + 255) / 256;
        egc_message_k<<<blocks, 256, 0, s>>>(w.Mb, w.G0, w.G1, src, dst, E);
    }
    hipMemsetAsync(w.SH, 0, (size_t)n * HID * sizeof(float), s);
    hipMemsetAsync(w.SS, 0, (size_t)n * HID * sizeof(float), s);
    {
        int blocks = (E * 32 + 255) / 256;
        egc_scatter_k<<<blocks, 256, 0, s>>>(w.SH, w.SS, w.Mb, w.G2, src, dst, E);
    }
    {
        int total = n * HID;
        egc_node_update_k<<<(total + 255) / 256, 256, 0, s>>>(
            x, w.G3, w.SH, w.SS, p.bn_n_g, p.bn_n_b, total);
    }
    {
        int total = E * HID;
        egc_edge_update_k<<<(total + 255) / 256, 256, 0, s>>>(
            y, w.Mb, p.bn_e_g, p.bn_e_b, total);
    }
}

} // namespace

extern "C" void kernel_launch(void* const* d_in, const int* in_sizes, int n_in,
                              void* d_out, int out_size, void* d_ws, size_t ws_size,
                              hipStream_t stream)
{
    (void)in_sizes; (void)n_in; (void)out_size; (void)ws_size;
    const int NN = 10000, NE = 120000, NLG = 720000;

    const float* atom   = (const float*)d_in[0];
    const float* rvec   = (const float*)d_in[1];
    const float* angleh = (const float*)d_in[2];
    const int*   g_src  = (const int*)d_in[3];
    const int*   g_dst  = (const int*)d_in[4];
    const int*   lg_src = (const int*)d_in[5];
    const int*   lg_dst = (const int*)d_in[6];

    // Params: JAX pytree order (sorted dict keys, recursive), base index 7.
    Egc aln_e0 = egc_at(d_in, 7);    // alignn[0].edge
    Egc aln_n0 = egc_at(d_in, 21);   // alignn[0].node
    Egc aln_e1 = egc_at(d_in, 35);   // alignn[1].edge
    Egc aln_n1 = egc_at(d_in, 49);   // alignn[1].node
    Mlp angle1 = mlp_at(d_in, 63);
    Mlp angle2 = mlp_at(d_in, 67);
    Mlp atomm  = mlp_at(d_in, 71);
    Mlp edge1  = mlp_at(d_in, 75);
    Mlp edge2  = mlp_at(d_in, 79);
    const float* fc_b = (const float*)d_in[83];
    const float* fc_w = (const float*)d_in[84];
    Egc gcn0 = egc_at(d_in, 85);
    Egc gcn1 = egc_at(d_in, 99);

    // Workspace layout (float offsets)
    float* ws = (float*)d_ws;
    Ws w;
    w.X  = ws;                  //  10000*128 =  1,280,000
    w.Y  = ws + 1280000;        // 120000*128 = 15,360,000
    w.Z  = ws + 16640000;       // 720000*128 = 92,160,000
    w.Mb = ws + 108800000;      // 720000*128 = 92,160,000
    w.G0 = ws + 200960000;      // 120000*128
    w.G1 = ws + 216320000;
    w.G2 = ws + 231680000;
    w.G3 = ws + 247040000;
    w.SH = ws + 262400000;
    w.SS = ws + 277760000;      // end: 293,120,000 floats (~1.1 GiB)

    // ---- Embeddings ----
    // x = mlp(atom_mlp, atom_features)
    launch_gemm(w.X, atom, atomm.w, atomm.b, NN, 92, HID, stream);
    launch_ln(w.X, atomm.ln_g, atomm.ln_b, NN, HID, stream);

    // y = mlp2(mlp1(rbf(|r|)))
    bond_rbf_k<<<(NE * 80 + 255) / 256, 256, 0, stream>>>(rvec, w.G0, NE);
    launch_gemm(w.G1, w.G0, edge1.w, edge1.b, NE, 80, 64, stream);
    launch_ln(w.G1, edge1.ln_g, edge1.ln_b, NE, 64, stream);
    launch_gemm(w.Y, w.G1, edge2.w, edge2.b, NE, 64, HID, stream);
    launch_ln(w.Y, edge2.ln_g, edge2.ln_b, NE, HID, stream);

    // z = mlp2(mlp1(rbf(angle_h)))  (transients packed disjointly inside Mb)
    float* rbf_a = w.Mb;                 // 720000*40 = 28.8M floats
    float* emb_a = w.Mb + 46080000;      // 720000*64 = 46.08M floats (disjoint)
    angle_rbf_k<<<(NLG * 40 + 255) / 256, 256, 0, stream>>>(angleh, rbf_a, NLG);
    launch_gemm(emb_a, rbf_a, angle1.w, angle1.b, NLG, 40, 64, stream);
    launch_ln(emb_a, angle1.ln_g, angle1.ln_b, NLG, 64, stream);
    launch_gemm(w.Z, emb_a, angle2.w, angle2.b, NLG, 64, HID, stream);
    launch_ln(w.Z, angle2.ln_g, angle2.ln_b, NLG, HID, stream);

    // ---- ALIGNN layers: node EGC on g, then edge EGC on line graph ----
    run_egc(aln_n0, g_src, g_dst, w.X, w.Y, NN, NE, w, stream);
    run_egc(aln_e0, lg_src, lg_dst, w.Y, w.Z, NE, NLG, w, stream);
    run_egc(aln_n1, g_src, g_dst, w.X, w.Y, NN, NE, w, stream);
    run_egc(aln_e1, lg_src, lg_dst, w.Y, w.Z, NE, NLG, w, stream);

    // ---- GCN layers ----
    run_egc(gcn0, g_src, g_dst, w.X, w.Y, NN, NE, w, stream);
    run_egc(gcn1, g_src, g_dst, w.X, w.Y, NN, NE, w, stream);

    // ---- Readout ----
    readout_k<<<1, HID, 0, stream>>>(w.X, fc_w, fc_b, (float*)d_out, NN);
}
